// RCAttention_34136400068766
// MI455X (gfx1250) — matrix operand order repaired and checked by reading, not compiled, not run on hardware
//
#include <hip/hip_runtime.h>
#include <hip/hip_bf16.h>

// CDNA5 / gfx1250, wave32. D = A(16x32 bf16) x B(32x16 bf16) + C(16x16 f32)
typedef __attribute__((ext_vector_type(16))) __bf16 v16bf;
typedef __attribute__((ext_vector_type(8)))  __bf16 v8bf;
typedef __attribute__((ext_vector_type(4)))  __bf16 v4bf;
typedef __attribute__((ext_vector_type(8)))  float  v8f;
typedef __attribute__((ext_vector_type(4)))  float  v4f;

#define HW    16384u     // H*W
#define HH    128
#define NCH   64
#define NELEM (16u * 64u * 16384u)
#define LDT   136        // bf16 LDS tile stride (mult of 8 -> 16B aligned rows)
#define LDS_S 132        // f32 LDS scratch stride
#define CST   72         // conv activation tile stride (pixel-major, 64ch + pad)

// ---- explicit global-address-space accessors (force global_load/store, not flat)
typedef __attribute__((address_space(1))) const v4f*   g_cv4f;
typedef __attribute__((address_space(1))) const v8bf*  g_cv8bf;
typedef __attribute__((address_space(1))) const float* g_cf;
typedef __attribute__((address_space(1))) v8bf*        g_v8bf;
typedef __attribute__((address_space(1))) v4f*         g_v4f;

__device__ __forceinline__ v4f   gload4f (const float*  p) { return *(g_cv4f)p; }
__device__ __forceinline__ v8bf  gload8bf(const __bf16* p) { return *(g_cv8bf)p; }
__device__ __forceinline__ float gloadf  (const float*  p) { return *(g_cf)p; }
__device__ __forceinline__ void  gstore8bf(__bf16* p, v8bf v) { *(g_v8bf)p = v; }
__device__ __forceinline__ void  gstore4f (float*  p, v4f  v) { *(g_v4f)p  = v; }

// ---- WMMA fragment loads, all 128-bit LDS reads ----------------------------
// A (16x32 bf16) from row-major LDS: lane l -> row m0+(l&15);
// elems are two contiguous 8-element k-runs at k0+8*(l>>4) and +16.
__device__ __forceinline__ v16bf ldA_rm(const __bf16* A, int m0, int k0,
                                        int sA, int lane) {
  const __bf16* p = A + (m0 + (lane & 15)) * sA + k0 + 8 * (lane >> 4);
  v8bf lo = *(const v8bf*)(p);
  v8bf hi = *(const v8bf*)(p + 16);
  return __builtin_shufflevector(lo, hi, 0, 1, 2, 3, 4, 5, 6, 7,
                                 8, 9, 10, 11, 12, 13, 14, 15);
}

// B (32x16 bf16) from column-major LDS (BT[n][k] row-major): lane l -> col
// n0+(l&15); elems are two contiguous 8-element k-runs at k0+8*(l>>4) and +16.
__device__ __forceinline__ v16bf ldB_ct(const __bf16* BT, int n0, int k0,
                                        int sB, int lane) {
  const __bf16* p = BT + (n0 + (lane & 15)) * sB + k0 + 8 * (lane >> 4);
  v8bf lo = *(const v8bf*)(p);
  v8bf hi = *(const v8bf*)(p + 16);
  return __builtin_shufflevector(lo, hi, 0, 1, 2, 3, 4, 5, 6, 7,
                                 8, 9, 10, 11, 12, 13, 14, 15);
}

// ---- 128x128x128 matmul: D = A * B; A row-major, B given column-major ------
// one wave per 16-row strip (8 waves cover the tile)
template <typename DT>
__device__ void mm128T(const __bf16* A, int sA, const __bf16* BT, int sB,
                       DT* D, int sD, int lane, int wave) {
  const int m0 = wave * 16, half = lane >> 4, nl = lane & 15;
  v16bf a[4];
#pragma unroll
  for (int kb = 0; kb < 4; ++kb) a[kb] = ldA_rm(A, m0, kb * 32, sA, lane);
#pragma unroll
  for (int nt = 0; nt < 8; ++nt) {
    v8f acc = {};
#pragma unroll
    for (int kb = 0; kb < 4; ++kb) {
      v16bf b = ldB_ct(BT, nt * 16, kb * 32, sB, lane);
      acc = __builtin_amdgcn_wmma_f32_16x16x32_bf16(
          false, a[kb], false, b, (short)0, acc, false, false);
    }
    const int col = nt * 16 + nl;
#pragma unroll
    for (int v = 0; v < 8; ++v)            // C/D: M = v + 8*half, N = nl
      D[(m0 + v + 8 * half) * sD + col] = (DT)acc[v];
  }
}

// ---- softmaxes over 128-wide rows of f32 LDS scratch -----------------------
__device__ void softmax_rm(const float* S, __bf16* D, int tid) {
  if (tid < HH) {
    const float* row = S + tid * LDS_S;
    float mx = -3.4e38f;
    for (int i = 0; i < HH; ++i) mx = fmaxf(mx, row[i]);
    float s = 0.f;
    for (int i = 0; i < HH; ++i) s += __expf(row[i] - mx);
    const float inv = 1.f / s;
    __bf16* dr = D + tid * LDT;            // row-major (A-operand next)
    for (int i = 0; i < HH; ++i) dr[i] = (__bf16)(__expf(row[i] - mx) * inv);
  }
}
__device__ void softmax_tr(const float* S, __bf16* DT_, int tid) {
  if (tid < HH) {
    const float* row = S + tid * LDS_S;
    float mx = -3.4e38f;
    for (int i = 0; i < HH; ++i) mx = fmaxf(mx, row[i]);
    float s = 0.f;
    for (int i = 0; i < HH; ++i) s += __expf(row[i] - mx);
    const float inv = 1.f / s;
    for (int i = 0; i < HH; ++i)           // transposed (B-operand next)
      DT_[i * LDT + tid] = (__bf16)(__expf(row[i] - mx) * inv);
  }
}

// ---- conv64 compute: D(64 x 16px) = W(64x64) @ X(64ch x 16px) + bias, ReLU -
// weights row-major in LDS (stride 64); activations pixel-major (XT, stride CST)
__device__ void conv64_wave(const __bf16* Wm, const __bf16* XT, const float* Bs,
                            __bf16* Os, int lane, int wave) {
  const int half = lane >> 4, nl = lane & 15;
  v16bf bx[2];
#pragma unroll
  for (int kb = 0; kb < 2; ++kb)
    bx[kb] = ldB_ct(XT, wave * 16, kb * 32, CST, lane);
#pragma unroll
  for (int mt = 0; mt < 4; ++mt) {
    v8f acc = {};
#pragma unroll
    for (int kb = 0; kb < 2; ++kb) {
      v16bf a = ldA_rm(Wm, mt * 16, kb * 32, NCH, lane);
      acc = __builtin_amdgcn_wmma_f32_16x16x32_bf16(
          false, a, false, bx[kb], (short)0, acc, false, false);
    }
#pragma unroll
    for (int v = 0; v < 8; ++v) {
      const int o = mt * 16 + v + 8 * half;
      Os[o * LDT + wave * 16 + nl] = (__bf16)fmaxf(acc[v] + Bs[o], 0.f);
    }
  }
}

// ---- kernels ---------------------------------------------------------------
__global__ void __launch_bounds__(256)
rcatt_conv_qkv(const float* __restrict__ x, const float* __restrict__ att,
               const float* __restrict__ w0, const float* __restrict__ b0,
               const float* __restrict__ w1, const float* __restrict__ b1,
               const float* __restrict__ w2, const float* __restrict__ b2,
               __bf16* __restrict__ Q, __bf16* __restrict__ K,
               __bf16* __restrict__ V) {
  extern __shared__ char smem[];
  __bf16* XT = (__bf16*)smem;              // 128px x CST  (x, pixel-major)
  __bf16* ATt = XT + HH * CST;             // att tile
  __bf16* W0s = ATt + HH * CST;            // 3 x 64x64 bf16 weights
  __bf16* W1s = W0s + NCH * NCH;
  __bf16* W2s = W1s + NCH * NCH;
  __bf16* Os  = W2s + NCH * NCH;           // 64 x LDT output staging
  float*  Bs  = (float*)(Os + NCH * LDT);  // 3 x 64 bias

  const int tid = threadIdx.x, lane = tid & 31, wave = tid >> 5;
  const int img = blockIdx.x >> 7;         // 128 blocks per image
  const int pix0 = (blockIdx.x & 127) << 7;
  const long base = (long)img * NCH * HW;

  // weights fp32 -> bf16 LDS (vectorized global_load_b128)
#pragma unroll
  for (int e = tid * 4; e < NCH * NCH; e += 1024) {
    v4f a0 = gload4f(w0 + e), a1 = gload4f(w1 + e), a2 = gload4f(w2 + e);
    v4bf p0, p1, p2;
#pragma unroll
    for (int j = 0; j < 4; ++j) { p0[j] = (__bf16)a0[j]; p1[j] = (__bf16)a1[j]; p2[j] = (__bf16)a2[j]; }
    *(v4bf*)&W0s[e] = p0; *(v4bf*)&W1s[e] = p1; *(v4bf*)&W2s[e] = p2;
  }
  if (tid < NCH) {
    Bs[tid]       = gloadf(b0 + tid);
    Bs[64 + tid]  = gloadf(b1 + tid);
    Bs[128 + tid] = gloadf(b2 + tid);
  }
  // activation tiles: x[c][pix0..+127], att -> pixel-major bf16 LDS
#pragma unroll
  for (int e = tid * 4; e < NCH * HH; e += 1024) {
    const int c = e >> 7, p = e & 127;
    v4f xv = gload4f(x   + base + (long)c * HW + pix0 + p);
    v4f av = gload4f(att + base + (long)c * HW + pix0 + p);
#pragma unroll
    for (int j = 0; j < 4; ++j) {
      XT [(p + j) * CST + c] = (__bf16)xv[j];
      ATt[(p + j) * CST + c] = (__bf16)av[j];
    }
  }
  __syncthreads();

#pragma unroll
  for (int ci = 0; ci < 3; ++ci) {
    const __bf16* Wm = (ci == 0) ? W0s : (ci == 1) ? W1s : W2s;
    const __bf16* Xin = (ci == 1) ? ATt : XT;
    conv64_wave(Wm, Xin, Bs + ci * 64, Os, lane, wave);
    __syncthreads();
    __bf16* dst = (ci == 0) ? Q : (ci == 1) ? K : V;
    for (int e = tid * 8; e < NCH * HH; e += 2048) {
      const int o = e >> 7, p = e & 127;
      gstore8bf(dst + base + (long)o * HW + pix0 + p, *(const v8bf*)&Os[o * LDT + p]);
    }
    __syncthreads();
  }
}

__global__ void __launch_bounds__(256)
rcatt_attention(const __bf16* __restrict__ Qg, const __bf16* __restrict__ Kg,
                const __bf16* __restrict__ Vg, __bf16* __restrict__ Og) {
  extern __shared__ char smem[];
  __bf16* Qs = (__bf16*)smem;      // Q row-major          (A of S1)
  __bf16* Ks = Qs + HH * LDT;      // K row-major = (K^T)c (B of S1); later A2^T
  __bf16* QT = Ks + HH * LDT;      // Q^T                  (B of S2)
  __bf16* KT = QT + HH * LDT;      // K^T row-major        (A of S2)
  __bf16* VT = KT + HH * LDT;      // V^T                  (B of T); later O stage
  __bf16* A1 = VT + HH * LDT;      // attn1 row-major      (A of T)
  float*  Sf = (float*)(A1 + HH * LDT);   // 128 x LDS_S f32 scratch

  const int tid = threadIdx.x, lane = tid & 31, wave = tid >> 5;
  const long base = (long)blockIdx.x * HW;   // one (b,c) slice per workgroup
  const __bf16* q = Qg + base;
  const __bf16* k = Kg + base;
  const __bf16* v = Vg + base;

  // vectorized load + build natural and transposed copies
  for (int e = tid * 8; e < (int)HW; e += 2048) {
    const int r = e >> 7, c = e & 127;
    v8bf qv = gload8bf(q + r * HH + c);
    v8bf kv = gload8bf(k + r * HH + c);
    v8bf vv = gload8bf(v + r * HH + c);
    *(v8bf*)&Qs[r * LDT + c] = qv;
    *(v8bf*)&Ks[r * LDT + c] = kv;
#pragma unroll
    for (int j = 0; j < 8; ++j) {
      QT[(c + j) * LDT + r] = qv[j];
      KT[(c + j) * LDT + r] = kv[j];
      VT[(c + j) * LDT + r] = vv[j];
    }
  }
  __syncthreads();

  mm128T(Qs, LDT, Ks, LDT, Sf, LDS_S, lane, wave);   // S1 = Q @ K^T
  __syncthreads();
  softmax_rm(Sf, A1, tid);                           // attn1 (row-major)
  __syncthreads();
  mm128T(KT, LDT, QT, LDT, Sf, LDS_S, lane, wave);   // S2 = K^T @ Q
  __syncthreads();
  softmax_tr(Sf, Ks, tid);                           // attn2^T -> reuse Ks
  __syncthreads();
  mm128T(A1, LDT, VT, LDT, Qs, LDT, lane, wave);     // T = attn1 @ V -> Qs
  __syncthreads();
  mm128T(Qs, LDT, Ks, LDT, VT, LDT, lane, wave);     // O = T @ attn2 -> VT
  __syncthreads();

  for (int e = tid * 8; e < (int)HW; e += 2048) {    // coalesced 16B stores
    const int r = e >> 7, c = e & 127;
    gstore8bf(Og + base + r * HH + c, *(const v8bf*)&VT[r * LDT + c]);
  }
}

__global__ void __launch_bounds__(256)
rcatt_conv_out(const __bf16* __restrict__ O, const float* __restrict__ w3,
               const float* __restrict__ b3, float* __restrict__ out) {
  extern __shared__ char smem[];
  __bf16* XT  = (__bf16*)smem;              // 128px x CST
  __bf16* W3s = XT + HH * CST;              // 64x64 bf16
  float*  OsF = (float*)(W3s + NCH * NCH);  // 64 x LDS_S f32 staging
  float*  Bs  = OsF + NCH * LDS_S;

  const int tid = threadIdx.x, lane = tid & 31, wave = tid >> 5;
  const int img = blockIdx.x >> 7;
  const int pix0 = (blockIdx.x & 127) << 7;
  const long base = (long)img * NCH * HW;

#pragma unroll
  for (int e = tid * 4; e < NCH * NCH; e += 1024) {
    v4f a = gload4f(w3 + e);
    v4bf pk;
#pragma unroll
    for (int j = 0; j < 4; ++j) pk[j] = (__bf16)a[j];
    *(v4bf*)&W3s[e] = pk;
  }
  if (tid < NCH) Bs[tid] = gloadf(b3 + tid);
  for (int e = tid * 8; e < NCH * HH; e += 2048) {
    const int c = e >> 7, p = e & 127;
    v8bf ov = gload8bf(O + base + (long)c * HW + pix0 + p);
#pragma unroll
    for (int j = 0; j < 8; ++j) XT[(p + j) * CST + c] = ov[j];
  }
  __syncthreads();

  // compute, staging output in f32
  {
    const int half = lane >> 4, nl = lane & 15;
    v16bf bx[2];
#pragma unroll
    for (int kb = 0; kb < 2; ++kb)
      bx[kb] = ldB_ct(XT, wave * 16, kb * 32, CST, lane);
#pragma unroll
    for (int mt = 0; mt < 4; ++mt) {
      v8f acc = {};
#pragma unroll
      for (int kb = 0; kb < 2; ++kb) {
        v16bf a = ldA_rm(W3s, mt * 16, kb * 32, NCH, lane);
        acc = __builtin_amdgcn_wmma_f32_16x16x32_bf16(
            false, a, false, bx[kb], (short)0, acc, false, false);
      }
#pragma unroll
      for (int vv = 0; vv < 8; ++vv) {
        const int o = mt * 16 + vv + 8 * half;
        OsF[o * LDS_S + wave * 16 + nl] = fmaxf(acc[vv] + Bs[o], 0.f);
      }
    }
  }
  __syncthreads();

  for (int e = tid * 4; e < NCH * HH; e += 1024) {   // coalesced 16B f32 stores
    const int o = e >> 7, p = e & 127;
    gstore4f(out + base + (long)o * HW + pix0 + p, *(const v4f*)&OsF[o * LDS_S + p]);
  }
}

// ---- host entry ------------------------------------------------------------
extern "C" void kernel_launch(void* const* d_in, const int* in_sizes, int n_in,
                              void* d_out, int out_size, void* d_ws, size_t ws_size,
                              hipStream_t stream) {
  const float* x   = (const float*)d_in[0];
  const float* att = (const float*)d_in[1];
  const float* w0  = (const float*)d_in[2];
  const float* b0  = (const float*)d_in[3];
  const float* w1  = (const float*)d_in[4];
  const float* b1  = (const float*)d_in[5];
  const float* w2  = (const float*)d_in[6];
  const float* b2  = (const float*)d_in[7];
  const float* w3  = (const float*)d_in[8];
  const float* b3  = (const float*)d_in[9];

  __bf16* Q = (__bf16*)d_ws;              // 4 bf16 tensors of B*C*H*W
  __bf16* K = Q + (size_t)NELEM;
  __bf16* V = K + (size_t)NELEM;
  __bf16* O = V + (size_t)NELEM;

  const size_t smem_qkv =
      (size_t)(2 * HH * CST + 3 * NCH * NCH + NCH * LDT) * sizeof(__bf16) +
      (size_t)(3 * NCH) * sizeof(float);
  const size_t smem_att =
      (size_t)(6 * HH * LDT) * sizeof(__bf16) +
      (size_t)(HH * LDS_S) * sizeof(float);
  const size_t smem_out =
      (size_t)(HH * CST + NCH * NCH) * sizeof(__bf16) +
      (size_t)(NCH * LDS_S + NCH) * sizeof(float);

  rcatt_conv_qkv<<<2048, 256, smem_qkv, stream>>>(x, att, w0, b0, w1, b1,
                                                  w2, b2, Q, K, V);
  rcatt_attention<<<1024, 256, smem_att, stream>>>(Q, K, V, O);
  rcatt_conv_out<<<2048, 256, smem_out, stream>>>(O, w3, b3, (float*)d_out);
}
